// SkipableGAT_52939766890864
// MI455X (gfx1250) — matrix operand order repaired and checked by reading, not compiled, not run on hardware
//
#include <hip/hip_runtime.h>

// ---------------------------------------------------------------------------
// SkipableGAT fused forward for gfx1250 (MI455X), wave32, WMMA bf16.
//   kernel 0 (wconv): fp32 weights -> bf16, transposed to [N][K] in d_ws
//   kernel 1 (frame_kernel): 2 frames per workgroup (halves L2 weight
//     re-reads, tightens WMMA M-padding), whole net kept in LDS.
//   512 threads = 16 waves; 160,000 B LDS -> 2 blocks / 320KB WGP.
//   Chained GEMMs write bf16 A-operands directly (no re-staging pass).
// ---------------------------------------------------------------------------

typedef __attribute__((ext_vector_type(16))) __bf16 v16bf;
typedef __attribute__((ext_vector_type(8)))  float  v8f;

#define NTHR   512   // 16 waves of 32
#define NWAVES (NTHR / 32)
#define FPB    2     // frames per block

// ---- edge lists (joint graph: 32 directed + 17 self; bone: 38 + 16 self) ----
__constant__ int c_JS[49] = {
  0,0,0,1,1,2,2,3,4,4,5,5,6,7,7,8,8,8,8,9,9,10,11,11,12,12,13,14,14,15,15,16,
  0,1,2,3,4,5,6,7,8,9,10,11,12,13,14,15,16};
__constant__ int c_JE[49] = {
  1,4,7,0,2,1,3,2,0,5,4,6,5,0,8,7,9,11,14,8,10,9,8,12,11,13,12,8,15,14,16,15,
  0,1,2,3,4,5,6,7,8,9,10,11,12,13,14,15,16};
__constant__ int c_BS[54] = {
  0,0,0,1,1,2,3,3,3,4,4,5,6,6,6,7,7,7,7,8,8,8,8,9,10,10,10,10,11,11,12,13,13,13,13,14,14,15,
  0,1,2,3,4,5,6,7,8,9,10,11,12,13,14,15};
__constant__ int c_BE[54] = {
  1,3,6,0,2,1,0,4,6,3,5,4,0,3,7,6,8,10,13,7,9,10,13,8,7,8,11,13,10,12,11,7,8,10,14,13,15,14,
  0,1,2,3,4,5,6,7,8,9,10,11,12,13,14,15};

struct GP {
  const float* x;
  const float *n1_g, *n1_b, *j_lin1_b, *j_a, *j_proj_b, *j_lin2_b, *ja1_b, *ja2_w, *ja2_b;
  const float *n2_g, *n2_b, *tb_w, *tb_b, *tj_w, *tj_b;
  const float *b_lin1_b, *b_a, *b_proj_b, *b_lin2_b, *ba1_b, *ba2_w, *ba2_b;
  const __bf16 *j_lin1t, *j_wqkvt, *j_projt, *j_lin2t, *ja1t;
  const __bf16 *b_lin1t, *b_wqkvt, *b_projt, *b_lin2t, *ba1t;
  float* out;
  int frames;
};

__device__ __forceinline__ float leaky(float v) { return v >= 0.f ? v : 0.2f * v; }

// Stage Mreal x K fp32 (row stride == K) into ceil16(Mreal) x K bf16 A buffer.
__device__ void stage_a(const float* src, int Mreal, int K, int kshift, __bf16* dst) {
  const int rows32 = (Mreal + 15) & ~15;
  for (int i = threadIdx.x; i < rows32 * K; i += NTHR) {
    const int r = i >> kshift, c = i & (K - 1);
    const float v = (r < Mreal) ? src[r * K + c] : 0.f;
    dst[i] = (__bf16)v;
  }
}

// Stage bf16(a+b), K fixed at 256 (gating MLP inputs).
__device__ void stage_a_sum(const float* a, const float* b, int Mreal, __bf16* dst) {
  const int rows32 = (Mreal + 15) & ~15;
  for (int i = threadIdx.x; i < rows32 * 256; i += NTHR) {
    const int r = i >> 8, c = i & 255;
    const float v = (r < Mreal) ? (a[r * 256 + c] + b[r * 256 + c]) : 0.f;
    dst[i] = (__bf16)v;
  }
}

// out[Mreal x N] = act(A(bf16, LDS) * Wt^T + bias); Wt is [N][K] bf16 in global.
// 16 waves cooperatively cover (ceil(M/16) x N/16) 16x16 WMMA tiles.
// A fragment (wave32, 16-bit 16x32): lane<16 -> K {0..7,16..23}, lane>=16 -> {8..15,24..31}
// B fragment (32x16): lane<16 -> K {0..7,16..23} of column (lane&15); lane>=16 -> {8..15,24..31}
// Full 16-row M tiles store unguarded; partial tiles are guarded, and with
// padZero they write bf16 zeros into pad rows (so outb can feed the next GEMM).
__device__ void gemm_wmma(const __bf16* __restrict__ Wt, const float* __restrict__ bias,
                          const __bf16* s_abf, int K, int N, int Mreal,
                          float* outf, __bf16* outb, int ldo, int act, int padZero) {
  const int lane    = threadIdx.x & 31;
  const int wave    = threadIdx.x >> 5;
  const int mtiles  = (Mreal + 15) >> 4;
  const int ntilesN = N >> 4;
  const int tiles   = mtiles * ntilesN;
  const int khalfA  = (lane >> 4) << 3;   // 0 or 8
  const int khalfB  = (lane >> 4) << 3;   // 0 or 8
  for (int t = wave; t < tiles; t += NWAVES) {
    const int mt = t / ntilesN;
    const int nt = t - mt * ntilesN;
    const __bf16* arowp = s_abf + (mt * 16 + (lane & 15)) * K;
    const __bf16* wcol  = Wt + (size_t)(nt * 16 + (lane & 15)) * K;
    v8f acc = {};
    for (int kk = 0; kk < K; kk += 32) {
      v16bf a, b;
      const __bf16* ap = arowp + kk + khalfA;
      ((uint4*)&a)[0] = *(const uint4*)(ap);
      ((uint4*)&a)[1] = *(const uint4*)(ap + 16);
      const __bf16* bp = wcol + kk + khalfB;
      ((uint4*)&b)[0] = *(const uint4*)(bp);
      ((uint4*)&b)[1] = *(const uint4*)(bp + 16);
      acc = __builtin_amdgcn_wmma_f32_16x16x32_bf16(false, a, false, b,
                                                    (short)0, acc, false, false);
    }
    // C/D layout: VGPR v, lane<16 -> (M=v, N=lane); lane>=16 -> (M=v+8, N=lane-16)
    const int orow0 = mt * 16 + ((lane >> 4) << 3);
    const int ocol  = nt * 16 + (lane & 15);
    const float bv  = bias ? bias[ocol] : 0.f;
    const bool full = (mt * 16 + 16) <= Mreal;
#pragma unroll
    for (int v = 0; v < 8; ++v) {
      const int r = orow0 + v;
      if (full || r < Mreal) {
        float val = acc[v] + bv;
        if (act == 1) val = fmaxf(val, 0.f);
        else if (act == 2) val = leaky(val);
        if (outb) outb[r * ldo + ocol] = (__bf16)val;
        else      outf[r * ldo + ocol] = val;
      } else if (padZero) {
        outb[r * ldo + ocol] = (__bf16)0.f;   // zero pad rows of chained A buffer
      }
    }
  }
}

// In-place LayerNorm over 256 channels, one wave per row.
__device__ void layernorm_rows(float* buf, int Mreal, const float* g, const float* b) {
  const int lane = threadIdx.x & 31;
  const int wave = threadIdx.x >> 5;
  for (int r = wave; r < Mreal; r += NWAVES) {
    float s = 0.f, s2 = 0.f;
    for (int c = lane; c < 256; c += 32) { const float v = buf[r * 256 + c]; s += v; s2 += v * v; }
    for (int off = 16; off > 0; off >>= 1) {
      s  += __shfl_down(s, off, 32);
      s2 += __shfl_down(s2, off, 32);
    }
    s  = __shfl(s, 0, 32);
    s2 = __shfl(s2, 0, 32);
    const float mean = s * (1.f / 256.f);
    const float var  = s2 * (1.f / 256.f) - mean * mean;
    const float rstd = rsqrtf(var + 1e-5f);
    for (int c = lane; c < 256; c += 32)
      buf[r * 256 + c] = (buf[r * 256 + c] - mean) * rstd * g[c] + b[c];
  }
}

// GAT edge attention + message scatter for nf frames.
// qkv (bf16): rows = f*nodes + n, 640 wide = [H=8][q32|k32|v16].
// msg out (fp32): [f*nodes + n][128]. LDS float atomics -> ds_add_f32.
__device__ void gat_aggregate(const __bf16* qkv, int nf, int E, const int* st, const int* en,
                              const float* avec, int nodes,
                              float* s_edge, float* s_node, float* s_red, float* msg) {
  const int tid = threadIdx.x;
  const int E8 = E * 8, E128 = E * 128;
  for (int i = tid; i < nf * nodes * 128; i += NTHR) msg[i] = 0.f;
  for (int i = tid; i < nf * nodes * 8; i += NTHR) s_node[i] = 0.f;
  __syncthreads();
  for (int idx = tid; idx < nf * E8; idx += NTHR) {
    const int fe = idx / E8, rem = idx - fe * E8;
    const int e = rem >> 3, h = rem & 7;
    const __bf16* q = qkv + (fe * nodes + st[e]) * 640 + h * 80;
    const __bf16* k = qkv + (fe * nodes + en[e]) * 640 + h * 80 + 32;
    float z = 0.f;
#pragma unroll
    for (int a = 0; a < 32; ++a) z += leaky((float)q[a] + (float)k[a]) * avec[a];
    s_edge[idx] = z;
  }
  __syncthreads();
  if (tid < nf * 8) {                       // global max over edges per (frame, head)
    const int fe = tid >> 3, h = tid & 7;
    float m = -3.0e38f;
    for (int e = 0; e < E; ++e) m = fmaxf(m, s_edge[fe * E8 + e * 8 + h]);
    s_red[tid] = m;
  }
  __syncthreads();
  for (int idx = tid; idx < nf * E8; idx += NTHR) {
    const int fe = idx / E8, rem = idx - fe * E8;
    const int e = rem >> 3, h = rem & 7;
    const float z = __expf(s_edge[idx] - s_red[fe * 8 + h]);
    s_edge[idx] = z;
    atomicAdd(&s_node[(fe * nodes + st[e]) * 8 + h], z);    // sigma scatter
  }
  __syncthreads();
  for (int idx = tid; idx < nf * E128; idx += NTHR) {
    const int fe = idx / E128, rem = idx - fe * E128;
    const int e = rem >> 7, r = rem & 127;
    const int h = r >> 4, d = r & 15;
    const float attn = s_edge[fe * E8 + e * 8 + h] /
                       s_node[(fe * nodes + st[e]) * 8 + h];
    const float val = attn * (float)qkv[(fe * nodes + en[e]) * 640 + h * 80 + 64 + d];
    atomicAdd(&msg[(fe * nodes + st[e]) * 128 + r], val);
  }
}

// s = softmax(leaky((h+x0)@W1 + b1) @ W2 + b2); dst = s0*x0 + s1*h  (dst may alias h)
__device__ void gate_blend(const float* h, const float* x0, int Mreal,
                           const __bf16* w1t, const float* b1,
                           const float* w2, const float* b2,
                           __bf16* s_abf, float* s_g, float* s_red, float* dst) {
  stage_a_sum(h, x0, Mreal, s_abf);
  __syncthreads();
  gemm_wmma(w1t, b1, s_abf, 256, 128, Mreal, s_g, nullptr, 128, 2, 0);
  __syncthreads();
  const int lane = threadIdx.x & 31;
  const int wave = threadIdx.x >> 5;
  for (int r = wave; r < Mreal; r += NWAVES) {
    float d0 = 0.f, d1 = 0.f;
    for (int c = lane; c < 128; c += 32) {
      const float gv = s_g[r * 128 + c];
      d0 += gv * w2[c * 2 + 0];
      d1 += gv * w2[c * 2 + 1];
    }
    for (int off = 16; off > 0; off >>= 1) {
      d0 += __shfl_down(d0, off, 32);
      d1 += __shfl_down(d1, off, 32);
    }
    if (lane == 0) {
      d0 += b2[0]; d1 += b2[1];
      const float m = fmaxf(d0, d1);
      const float e0 = __expf(d0 - m), e1 = __expf(d1 - m);
      const float inv = 1.f / (e0 + e1);
      s_red[r * 2 + 0] = e0 * inv;
      s_red[r * 2 + 1] = e1 * inv;
    }
  }
  __syncthreads();
  for (int i = threadIdx.x; i < Mreal * 256; i += NTHR) {
    const int r = i >> 8;
    dst[i] = s_red[r * 2] * x0[i] + s_red[r * 2 + 1] * h[i];
  }
}

__global__ __launch_bounds__(NTHR) void frame_kernel(GP p) {
  // 160,000 B static LDS: two blocks co-resident per 320KB WGP.
  __shared__ float  s_x0[FPB * 17 * 256];              // 34816: LN'd input (skips)
  __shared__ __align__(16) unsigned char s_pool[FPB * 17 * 640 * 2]; // 43520: qkv(bf16)/fp32 scratch
  __shared__ float  s_h[FPB * 17 * 256];               // 34816: 256-wide activations
  __shared__ float  s_g[FPB * 17 * 128];               // 17408: 128-wide acts / msgs / bf16 A-chain
  __shared__ __bf16 s_abf[48 * 256];                   // 24576: padded WMMA A operand
  __shared__ float  s_edge[FPB * 54 * 8];              // 3456
  __shared__ float  s_node[FPB * 17 * 8];              // 1088
  __shared__ float  s_red[80];                         // 320

  float*  pool_f = (float*)s_pool;
  __bf16* pool_b = (__bf16*)s_pool;
  __bf16* s_gb   = (__bf16*)s_g;        // 48x128 bf16 chained-A overlay (12288 <= 17408 B)

  const int f0 = blockIdx.x * FPB;
  const int nf = min(FPB, p.frames - f0);
  const int rows_j = nf * 17;           // 34 (or 17 on a tail block)
  const int rows_b = nf * 16;           // 32 -> bone GEMMs have no partial M tile

  if (threadIdx.x == 0) {               // pull weight stream toward this WGP/L2
    __builtin_prefetch(p.j_lin1t, 0, 1);
    __builtin_prefetch(p.j_wqkvt, 0, 1);
  }
  const float* xin = p.x + (size_t)f0 * 17 * 256;
  for (int i = threadIdx.x; i < rows_j * 256; i += NTHR) s_x0[i] = xin[i];
  __syncthreads();
  layernorm_rows(s_x0, rows_j, p.n1_g, p.n1_b);
  __syncthreads();

  // ---------------- joint branch ----------------
  stage_a(s_x0, rows_j, 256, 8, s_abf);                                        __syncthreads();
  gemm_wmma(p.j_lin1t, p.j_lin1_b, s_abf, 256, 128, rows_j, nullptr, s_gb, 128, 0, 1);
  __syncthreads();
  gemm_wmma(p.j_wqkvt, nullptr, s_gb, 128, 640, rows_j, nullptr, pool_b, 640, 0, 0);
  __syncthreads();
  gat_aggregate(pool_b, nf, 49, c_JS, c_JE, p.j_a, 17, s_edge, s_node, s_red, s_g);
  __syncthreads();
  stage_a(s_g, rows_j, 128, 7, s_abf);                                         __syncthreads();
  gemm_wmma(p.j_projt, p.j_proj_b, s_abf, 128, 128, rows_j, nullptr, s_gb, 128, 1, 1);
  __syncthreads();
  gemm_wmma(p.j_lin2t, p.j_lin2_b, s_gb, 128, 256, rows_j, s_h, nullptr, 256, 0, 0);
  __syncthreads();
  gate_blend(s_h, s_x0, rows_j, p.ja1t, p.ja1_b, p.ja2_w, p.ja2_b, s_abf, s_g, s_red, s_h);
  __syncthreads();

  // joints -> bones mixing (17 -> 16), xb into fp32 pool (qkv is dead)
  for (int i = threadIdx.x; i < rows_b * 256; i += NTHR) {
    const int f = i >> 12, kb = (i >> 8) & 15, c = i & 255;
    float acc = p.tb_b[kb];
    for (int j = 0; j < 17; ++j) acc += s_h[(f * 17 + j) * 256 + c] * p.tb_w[j * 16 + kb];
    pool_f[i] = acc;
  }
  __syncthreads();
  layernorm_rows(pool_f, rows_b, p.n2_g, p.n2_b);
  __syncthreads();

  // ---------------- bone branch ----------------
  stage_a(pool_f, rows_b, 256, 8, s_abf);                                      __syncthreads();
  gemm_wmma(p.b_lin1t, p.b_lin1_b, s_abf, 256, 128, rows_b, nullptr, s_gb, 128, 0, 1);
  __syncthreads();
  gemm_wmma(p.b_wqkvt, nullptr, s_gb, 128, 640, rows_b, nullptr, pool_b, 640, 0, 0);
  __syncthreads();
  gat_aggregate(pool_b, nf, 54, c_BS, c_BE, p.b_a, 16, s_edge, s_node, s_red, s_g);
  __syncthreads();
  stage_a(s_g, rows_b, 128, 7, s_abf);                                         __syncthreads();
  gemm_wmma(p.b_projt, p.b_proj_b, s_abf, 128, 128, rows_b, nullptr, s_gb, 128, 1, 1);
  __syncthreads();
  gemm_wmma(p.b_lin2t, p.b_lin2_b, s_gb, 128, 256, rows_b, s_h, nullptr, 256, 0, 0);
  __syncthreads();

  // bones -> joints mixing (16 -> 17) into fp32 pool (qkv is dead)
  for (int i = threadIdx.x; i < rows_j * 256; i += NTHR) {
    const int f = i / 4352, rem = i - f * 4352;
    const int j = rem >> 8, c = rem & 255;
    float acc = p.tj_b[j];
    for (int k = 0; k < 16; ++k) acc += s_h[(f * 16 + k) * 256 + c] * p.tj_w[k * 17 + j];
    pool_f[i] = acc;
  }
  __syncthreads();

  gate_blend(pool_f, s_x0, rows_j, p.ba1t, p.ba1_b, p.ba2_w, p.ba2_b, s_abf, s_g, s_red,
             p.out + (size_t)f0 * 17 * 256);
}

// fp32 [K][N] -> bf16 transposed [N][K]
__global__ void wconv(const float* __restrict__ src, __bf16* __restrict__ dst, int K, int N) {
  const int i = blockIdx.x * blockDim.x + threadIdx.x;
  if (i < K * N) {
    const int k = i / N, n = i - k * N;
    dst[n * K + k] = (__bf16)src[i];
  }
}

// bf16 workspace layout (element offsets)
#define OFF_JL1  0            // 128x256
#define OFF_JQKV 32768        // 640x128
#define OFF_JPRJ 114688       // 128x128
#define OFF_JL2  131072       // 256x128
#define OFF_JA1  163840       // 128x256
#define OFF_BL1  196608
#define OFF_BQKV 229376
#define OFF_BPRJ 311296
#define OFF_BL2  327680
#define OFF_BA1  360448       // total 393216 bf16 = 768 KiB

extern "C" void kernel_launch(void* const* d_in, const int* in_sizes, int n_in,
                              void* d_out, int out_size, void* d_ws, size_t ws_size,
                              hipStream_t stream) {
  (void)n_in; (void)out_size; (void)ws_size;
  const float* F[33];
  for (int i = 0; i < 33; ++i) F[i] = (const float*)d_in[i];
  // 0:x 1:n1_g 2:n1_b 3:j_lin1_w 4:j_lin1_b 5:j_wqkv 6:j_a 7:j_proj_w 8:j_proj_b
  // 9:j_lin2_w 10:j_lin2_b 11:ja1_w 12:ja1_b 13:ja2_w 14:ja2_b 15:n2_g 16:n2_b
  // 17:tb_w 18:tb_b 19:tj_w 20:tj_b 21:b_lin1_w 22:b_lin1_b 23:b_wqkv 24:b_a
  // 25:b_proj_w 26:b_proj_b 27:b_lin2_w 28:b_lin2_b 29:ba1_w 30:ba1_b 31:ba2_w 32:ba2_b
  __bf16* ws = (__bf16*)d_ws;
  auto wc = [&](const float* s, __bf16* d, int K, int N) {
    const int tot = K * N;
    wconv<<<(tot + 255) / 256, 256, 0, stream>>>(s, d, K, N);
  };
  wc(F[3],  ws + OFF_JL1,  256, 128);
  wc(F[5],  ws + OFF_JQKV, 128, 640);
  wc(F[7],  ws + OFF_JPRJ, 128, 128);
  wc(F[9],  ws + OFF_JL2,  128, 256);
  wc(F[11], ws + OFF_JA1,  256, 128);
  wc(F[21], ws + OFF_BL1,  256, 128);
  wc(F[23], ws + OFF_BQKV, 128, 640);
  wc(F[25], ws + OFF_BPRJ, 128, 128);
  wc(F[27], ws + OFF_BL2,  128, 256);
  wc(F[29], ws + OFF_BA1,  256, 128);

  GP p;
  p.x = F[0];
  p.n1_g = F[1];  p.n1_b = F[2];  p.j_lin1_b = F[4];  p.j_a = F[6];
  p.j_proj_b = F[8]; p.j_lin2_b = F[10]; p.ja1_b = F[12]; p.ja2_w = F[13]; p.ja2_b = F[14];
  p.n2_g = F[15]; p.n2_b = F[16]; p.tb_w = F[17]; p.tb_b = F[18]; p.tj_w = F[19]; p.tj_b = F[20];
  p.b_lin1_b = F[22]; p.b_a = F[24]; p.b_proj_b = F[26]; p.b_lin2_b = F[28];
  p.ba1_b = F[30]; p.ba2_w = F[31]; p.ba2_b = F[32];
  p.j_lin1t = ws + OFF_JL1;  p.j_wqkvt = ws + OFF_JQKV; p.j_projt = ws + OFF_JPRJ;
  p.j_lin2t = ws + OFF_JL2;  p.ja1t    = ws + OFF_JA1;
  p.b_lin1t = ws + OFF_BL1;  p.b_wqkvt = ws + OFF_BQKV; p.b_projt = ws + OFF_BPRJ;
  p.b_lin2t = ws + OFF_BL2;  p.ba1t    = ws + OFF_BA1;
  p.out = (float*)d_out;
  p.frames = in_sizes[0] / (17 * 256);   // B*T = 3888

  const int blocks = (p.frames + FPB - 1) / FPB;
  frame_kernel<<<blocks, NTHR, 0, stream>>>(p);
}
